// CombinedModel_90460601188993
// MI455X (gfx1250) — matrix operand order repaired and checked by reading, not compiled, not run on hardware
//
#include <hip/hip_runtime.h>
#include <hip/hip_bf16.h>
#include <math.h>

// ---- problem constants (match reference) ----
#define BB 64
#define CC 64
#define TT 4096

// LDS row pitch in halfs: 64 data + 8 pad -> 144 bytes (16B aligned, spreads banks)
#define PITCH 72

typedef _Float16 h8   __attribute__((ext_vector_type(8)));
typedef _Float16 v16h __attribute__((ext_vector_type(16)));
typedef float    v8f  __attribute__((ext_vector_type(8)));

#if defined(__has_builtin)
#if __has_builtin(__builtin_amdgcn_tanhf)
#define TANHF(x) __builtin_amdgcn_tanhf(x)
#endif
#endif
#ifndef TANHF
#define TANHF(x) tanhf(x)
#endif

#define CAT16(lo, hh) __builtin_shufflevector(lo, hh, 0,1,2,3,4,5,6,7,8,9,10,11,12,13,14,15)

// One workgroup = one 16-row batch tile, 4 waves = four 16-col N-tiles.
// State x kept in fp32 (accumulator layout) in registers; tanh(x) and x
// mirrored into LDS as f16 A-operands for the two WMMAs per step.
__global__ __launch_bounds__(128, 1)
void rnode_scan_kernel(const float* __restrict__ u,
                       const float* __restrict__ W_rec,
                       const float* __restrict__ W_ff,
                       const float* __restrict__ b_ff,
                       const float* __restrict__ dtp,
                       float* __restrict__ out_y,
                       float* __restrict__ out_x)
{
    __shared__ __align__(16) _Float16 sh_x[16 * PITCH]; // x_t  (f16 copy)
    __shared__ __align__(16) _Float16 sh_t[16 * PITCH]; // tanh(x_t) (f16)

    const int tid      = threadIdx.x;
    const int lane     = tid & 31;
    const int wave     = tid >> 5;       // N-tile 0..3
    const int hi       = lane >> 4;      // half-wave select
    const int ln       = lane & 15;
    const int col_base = wave * 16;
    const int bbase    = blockIdx.x * 16;

    const float dt       = dtp[0];
    const float one_m_dt = 1.0f - dt;

    // ---- preload time-invariant weights as f16 WMMA B-fragments ----
    // B[k][n] = W[n][k];  element j of fragment c = W[n][c*32 + 8*hi + j + (j>=8 ? 8 : 0)]
    v16h brec[2], bff[2];
    {
        const int nrow = col_base + ln;
        #pragma unroll
        for (int c = 0; c < 2; ++c) {
            const float* wr = &W_rec[nrow * CC + c * 32 + 8 * hi];
            const float* wf = &W_ff [nrow * CC + c * 32 + 8 * hi];
            #pragma unroll
            for (int j = 0; j < 16; ++j) {
                brec[c][j] = (_Float16)wr[j + (j >= 8 ? 8 : 0)];
                bff [c][j] = (_Float16)wf[j + (j >= 8 ? 8 : 0)];
            }
        }
    }
    const float bias = b_ff[col_base + ln];

    // zero-init LDS (x0 = 0 -> tanh(x0) = 0)
    for (int i = tid; i < 16 * PITCH; i += 128) {
        sh_x[i] = (_Float16)0.0f;
        sh_t[i] = (_Float16)0.0f;
    }
    __syncthreads();

    // this lane owns 8 accumulator-layout elements: m = i + 8*hi, n = col_base+ln
    const long rowstep = (long)CC * TT; // elements between batch rows in [B,C,T]
    const long gbase   = ((long)(bbase + 8 * hi) * CC + col_base + ln) * TT;
    const float* up = u     + gbase;
    float*       yp = out_y + gbase;
    float*       xp = out_x + gbase;

    float xreg[8];
    #pragma unroll
    for (int i = 0; i < 8; ++i) xreg[i] = 0.0f;

    // A-operand read bases (lane = matrix row), write bases (accumulator layout)
    const _Float16* tA = &sh_t[ln * PITCH + 8 * hi];
    const _Float16* xA = &sh_x[ln * PITCH + 8 * hi];
    _Float16* xW = &sh_x[(8 * hi) * PITCH + col_base + ln];
    _Float16* tW = &sh_t[(8 * hi) * PITCH + col_base + ln];

    for (int t = 0; t < TT; ++t) {
        // ---- GEMM1: acc = tanh(x_{t-1}) @ W_rec^T  (critical path) ----
        h8 l0 = *(const h8*)(tA + 0);
        h8 h0 = *(const h8*)(tA + 16);
        h8 l1 = *(const h8*)(tA + 32);
        h8 h1 = *(const h8*)(tA + 48);
        v16h a0 = CAT16(l0, h0);
        v16h a1 = CAT16(l1, h1);

        v8f acc = {0.f, 0.f, 0.f, 0.f, 0.f, 0.f, 0.f, 0.f};
        acc = __builtin_amdgcn_wmma_f32_16x16x32_f16(false, a0, false, brec[0],
                                                     (short)0, acc, false, false);
        acc = __builtin_amdgcn_wmma_f32_16x16x32_f16(false, a1, false, brec[1],
                                                     (short)0, acc, false, false);

        // drive input; prefetch the next 128B line of each u row once per 32 steps
        float uf[8];
        #pragma unroll
        for (int i = 0; i < 8; ++i) uf[i] = up[(long)i * rowstep + t];
        if ((t & 31) == 0 && t + 32 < TT) {
            #pragma unroll
            for (int i = 0; i < 8; ++i)
                __builtin_prefetch(&up[(long)i * rowstep + t + 32], 0, 1);
        }

        __syncthreads(); // all waves finished reading sh_t/sh_x of prev step

        // ---- Euler update (fp32), mirror x_t & tanh(x_t) to LDS as f16 ----
        #pragma unroll
        for (int i = 0; i < 8; ++i) {
            float xn = one_m_dt * xreg[i] + dt * (acc[i] + uf[i]);
            xreg[i]  = xn;
            float tn = TANHF(xn);
            xW[i * PITCH] = (_Float16)xn;
            tW[i * PITCH] = (_Float16)tn;
            xp[(long)i * rowstep + t] = xn;   // membrane potential output
        }
        __syncthreads(); // x_t / tanh(x_t) visible to all waves

        // ---- GEMM2 (readout, off critical path): y = x_t @ W_ff^T + b ----
        h8 m0 = *(const h8*)(xA + 0);
        h8 n0 = *(const h8*)(xA + 16);
        h8 m1 = *(const h8*)(xA + 32);
        h8 n1 = *(const h8*)(xA + 48);
        v16h c0 = CAT16(m0, n0);
        v16h c1 = CAT16(m1, n1);

        v8f acc2;
        #pragma unroll
        for (int i = 0; i < 8; ++i) acc2[i] = bias;
        acc2 = __builtin_amdgcn_wmma_f32_16x16x32_f16(false, c0, false, bff[0],
                                                      (short)0, acc2, false, false);
        acc2 = __builtin_amdgcn_wmma_f32_16x16x32_f16(false, c1, false, bff[1],
                                                      (short)0, acc2, false, false);

        #pragma unroll
        for (int i = 0; i < 8; ++i)
            yp[(long)i * rowstep + t] = acc2[i];
    }
}

extern "C" void kernel_launch(void* const* d_in, const int* in_sizes, int n_in,
                              void* d_out, int out_size, void* d_ws, size_t ws_size,
                              hipStream_t stream) {
    (void)in_sizes; (void)n_in; (void)out_size; (void)d_ws; (void)ws_size;
    const float* u     = (const float*)d_in[0];   // [B, C, T]
    const float* W_rec = (const float*)d_in[1];   // [C, C]
    const float* W_ff  = (const float*)d_in[2];   // [C, C]
    const float* b_ff  = (const float*)d_in[3];   // [C]
    const float* dt    = (const float*)d_in[4];   // scalar

    float* out_y = (float*)d_out;                     // outputs            [B,C,T]
    float* out_x = (float*)d_out + (long)BB * CC * TT; // membrane potentials [B,C,T]

    dim3 grid(BB / 16);   // 4 batch tiles
    dim3 block(128);      // 4 waves: one per 16-column N-tile
    rnode_scan_kernel<<<grid, block, 0, stream>>>(u, W_rec, W_ff, b_ff, dt,
                                                  out_y, out_x);
}
